// AtomicRouteConv_6270652252798
// MI455X (gfx1250) — matrix operand order repaired and checked by reading, not compiled, not run on hardware
//
#include <hip/hip_runtime.h>
#include <math.h>

typedef __attribute__((ext_vector_type(16))) __bf16 v16bf;
typedef __attribute__((ext_vector_type(8)))  __bf16 v8bf;
typedef __attribute__((ext_vector_type(8)))  float  v8f;

#define C128 128

// ---------------------------------------------------------------------------
// small helpers
// ---------------------------------------------------------------------------
__device__ __forceinline__ void atomic_max_f32(float* addr, float v) {
  // order-preserving bit trick; gmax is initialized to 0xFF800000 (-inf)
  if (v >= 0.0f) atomicMax((int*)addr, __float_as_int(v));
  else           atomicMin((unsigned int*)addr, __float_as_uint(v));
}

__global__ void k_zero_f32(float* __restrict__ p, long long n) {
  long long i = (long long)blockIdx.x * blockDim.x + threadIdx.x;
  long long n4 = n >> 2;
  if (i < n4) {
    float4 z; z.x = 0.f; z.y = 0.f; z.z = 0.f; z.w = 0.f;
    ((float4*)p)[i] = z;
  }
  if (i < (n & 3)) p[(n4 << 2) + i] = 0.0f;   // ragged tail (unused at these sizes)
}

__global__ void k_init_small(const float* __restrict__ b1, const float* __restrict__ b2,
                             float* __restrict__ bsum, float* __restrict__ gmax,
                             float* __restrict__ Z) {
  int i = threadIdx.x;
  if (i < C128) bsum[i] = b1[i] + b2[i];
  if (i == 0) { *gmax = __int_as_float(0xFF800000); *Z = 0.0f; }
}

__global__ void k_cvt_bf16(const float* __restrict__ W, __bf16* __restrict__ O, int n) {
  int i = blockIdx.x * blockDim.x + threadIdx.x;
  if (i < n) O[i] = (__bf16)W[i];
}

// ---------------------------------------------------------------------------
// stage-1 edge scatter: agg[m,:] += x_src[s,:]; deg[m] += 1   (wave per edge)
// ---------------------------------------------------------------------------
__global__ void k_scatter1(const int* __restrict__ ei, const float* __restrict__ xsrc,
                           float* __restrict__ agg, float* __restrict__ deg, int E) {
  int lane = threadIdx.x & 31;
  long long w = ((long long)blockIdx.x * blockDim.x + threadIdx.x) >> 5;
  if (w >= E) return;                    // uniform per wave
  int e = (int)w;
  int s = ei[e];                         // row 0: src
  int m = ei[E + e];                     // row 1: mid
  const float4 xv = ((const float4*)(xsrc + (long long)s * C128))[lane];
  float* ap = agg + (long long)m * C128 + lane * 4;
  atomicAdd(ap + 0, xv.x);
  atomicAdd(ap + 1, xv.y);
  atomicAdd(ap + 2, xv.z);
  atomicAdd(ap + 3, xv.w);
  if (lane == 0) atomicAdd(deg + m, 1.0f);
}

// ---------------------------------------------------------------------------
// WMMA fragment loaders (layouts per CDNA5 ISA 7.12.2)
// A 16x32 bf16: lane L, row = r0 + (L&15); hi = L>>4
//   element i<8  -> K = 32k + 8*hi + i
//   element i>=8 -> K = 32k + 16 + 8*hi + (i-8)
// B 32x16 bf16 (B = W^T): lane L, col c = nt*16 + (L&15)
//   element i<8 -> K = 32k + 8*hi + i; i>=8 -> K = 32k + 16 + 8*hi + (i-8)
// C/D 16x16 f32: lane L, VGPR g -> row r0 + g + 8*hi, col = nt*16 + (L&15)
// ---------------------------------------------------------------------------
__device__ __forceinline__ v16bf load_afrag(const float* __restrict__ xr, int k, int hi) {
  const float* p0 = xr + 32 * k + 8 * hi;
  const float* p1 = p0 + 16;
  const float4 f0 = *(const float4*)(p0);
  const float4 f1 = *(const float4*)(p0 + 4);
  const float4 f2 = *(const float4*)(p1);
  const float4 f3 = *(const float4*)(p1 + 4);
  v16bf a;
  a[0]  = (__bf16)f0.x; a[1]  = (__bf16)f0.y; a[2]  = (__bf16)f0.z; a[3]  = (__bf16)f0.w;
  a[4]  = (__bf16)f1.x; a[5]  = (__bf16)f1.y; a[6]  = (__bf16)f1.z; a[7]  = (__bf16)f1.w;
  a[8]  = (__bf16)f2.x; a[9]  = (__bf16)f2.y; a[10] = (__bf16)f2.z; a[11] = (__bf16)f2.w;
  a[12] = (__bf16)f3.x; a[13] = (__bf16)f3.y; a[14] = (__bf16)f3.z; a[15] = (__bf16)f3.w;
  return a;
}

__device__ __forceinline__ v16bf load_bfrag(const __bf16* __restrict__ Wb, int c, int k, int hi) {
  const __bf16* wp = Wb + c * C128 + 32 * k + 8 * hi;   // 16B aligned
  v8bf lo = *(const v8bf*)(wp);
  v8bf hh = *(const v8bf*)(wp + 16);
  return __builtin_shufflevector(lo, hh, 0,1,2,3,4,5,6,7,8,9,10,11,12,13,14,15);
}

// ---------------------------------------------------------------------------
// generic Y = X @ W^T + b     (X: N x 128 f32, W: 128x128 bf16 row-major[c][j])
// one wave per 16-row strip, 8 waves (128 rows) per block
// ---------------------------------------------------------------------------
__global__ void k_gemm_bias(const float* __restrict__ X, const __bf16* __restrict__ Wb,
                            const float* __restrict__ bias, float* __restrict__ Y, int N) {
  const int lane = threadIdx.x & 31;
  const int wave = threadIdx.x >> 5;
  const int hi = lane >> 4, lr = lane & 15;
  int r0 = (blockIdx.x * 8 + wave) * 16;
  if (r0 >= N) return;                   // uniform per wave
  const bool fullTile = (r0 + 16 <= N);  // uniform per wave
  int mrow = r0 + lr; if (mrow >= N) mrow = N - 1;
  const float* xr = X + (long long)mrow * C128;

  v16bf af[4];
#pragma unroll
  for (int k = 0; k < 4; ++k) af[k] = load_afrag(xr, k, hi);

#pragma unroll
  for (int nt = 0; nt < 8; ++nt) {
    int c = nt * 16 + lr;
    v8f acc = {0.f,0.f,0.f,0.f,0.f,0.f,0.f,0.f};
#pragma unroll
    for (int k = 0; k < 4; ++k) {
      v16bf bf = load_bfrag(Wb, c, k, hi);
      acc = __builtin_amdgcn_wmma_f32_16x16x32_bf16(false, af[k], false, bf,
                                                    (short)0, acc, false, false);
    }
    float bv = bias[c];
    float* yp = Y + (long long)(r0 + 8 * hi) * C128 + c;
    if (fullTile) {
#pragma unroll
      for (int g = 0; g < 8; ++g) yp[(long long)g * C128] = acc[g] + bv;
    } else {
#pragma unroll
      for (int g = 0; g < 8; ++g) {
        int row = r0 + g + 8 * hi;
        if (row < N) Y[(long long)row * C128 + c] = acc[g] + bv;
      }
    }
  }
}

// ---------------------------------------------------------------------------
// stage-1 node GEMM:  H = deg * (Xmid @ W1^T) + Agg @ W2^T + deg * (b1 + b2)
// ---------------------------------------------------------------------------
__global__ void k_gemm_mid(const float* __restrict__ Xmid, const float* __restrict__ Agg,
                           const __bf16* __restrict__ W1b, const __bf16* __restrict__ W2b,
                           const float* __restrict__ deg, const float* __restrict__ bsum,
                           float* __restrict__ H, int N) {
  const int lane = threadIdx.x & 31;
  const int wave = threadIdx.x >> 5;
  const int hi = lane >> 4, lr = lane & 15;
  int r0 = (blockIdx.x * 8 + wave) * 16;
  if (r0 >= N) return;
  const bool fullTile = (r0 + 16 <= N);
  int mrow = r0 + lr; if (mrow >= N) mrow = N - 1;
  const float* xr = Xmid + (long long)mrow * C128;
  const float* ar = Agg  + (long long)mrow * C128;

  v16bf am[4], aa[4];
#pragma unroll
  for (int k = 0; k < 4; ++k) { am[k] = load_afrag(xr, k, hi); aa[k] = load_afrag(ar, k, hi); }

  float dg[8];
#pragma unroll
  for (int g = 0; g < 8; ++g) {
    int row = r0 + g + 8 * hi; if (row >= N) row = N - 1;
    dg[g] = deg[row];
  }

#pragma unroll
  for (int nt = 0; nt < 8; ++nt) {
    int c = nt * 16 + lr;
    v8f acc = {0.f,0.f,0.f,0.f,0.f,0.f,0.f,0.f};
#pragma unroll
    for (int k = 0; k < 4; ++k) {
      v16bf bf = load_bfrag(W1b, c, k, hi);
      acc = __builtin_amdgcn_wmma_f32_16x16x32_bf16(false, am[k], false, bf,
                                                    (short)0, acc, false, false);
    }
    float bs = bsum[c];
#pragma unroll
    for (int g = 0; g < 8; ++g) acc[g] = dg[g] * (acc[g] + bs);   // deg*(x@W1^T + b1+b2)
#pragma unroll
    for (int k = 0; k < 4; ++k) {
      v16bf bf = load_bfrag(W2b, c, k, hi);
      acc = __builtin_amdgcn_wmma_f32_16x16x32_bf16(false, aa[k], false, bf,
                                                    (short)0, acc, false, false);
    }
    float* hp = H + (long long)(r0 + 8 * hi) * C128 + c;
    if (fullTile) {
#pragma unroll
      for (int g = 0; g < 8; ++g) hp[(long long)g * C128] = acc[g];
    } else {
#pragma unroll
      for (int g = 0; g < 8; ++g) {
        int row = r0 + g + 8 * hi;
        if (row < N) H[(long long)row * C128 + c] = acc[g];
      }
    }
  }
}

// ---------------------------------------------------------------------------
// per-edge attention scores + global max  (thread per edge)
// ---------------------------------------------------------------------------
__global__ void k_scores(const int* __restrict__ ei2, const float* __restrict__ Q,
                         const float* __restrict__ K, float* __restrict__ s,
                         float* __restrict__ gmax, int E) {
  __shared__ float wred[8];
  long long e = (long long)blockIdx.x * blockDim.x + threadIdx.x;
  float sc = -3.0e38f;
  if (e < E) {
    int m = ei2[e];          // row 0: mid
    int d = ei2[E + e];      // row 1: dst
    const float4* qp = (const float4*)(Q + (long long)d * C128);
    const float4* kp = (const float4*)(K + (long long)m * C128);
    float acc = 0.f;
#pragma unroll
    for (int t = 0; t < 32; ++t) {
      float4 a = qp[t], b = kp[t];
      acc = fmaf(a.x, b.x, acc); acc = fmaf(a.y, b.y, acc);
      acc = fmaf(a.z, b.z, acc); acc = fmaf(a.w, b.w, acc);
    }
    sc = acc * 0.08838834764831845f;    // 1/sqrt(128)
    s[e] = sc;
  }
#pragma unroll
  for (int off = 16; off > 0; off >>= 1) sc = fmaxf(sc, __shfl_xor(sc, off, 32));
  if ((threadIdx.x & 31) == 0) wred[threadIdx.x >> 5] = sc;
  __syncthreads();
  if (threadIdx.x == 0) {
    float v = wred[0];
#pragma unroll
    for (int i = 1; i < 8; ++i) v = fmaxf(v, wred[i]);
    atomic_max_f32(gmax, v);
  }
}

// ---------------------------------------------------------------------------
// global sum of exp(s - gmax)
// ---------------------------------------------------------------------------
__global__ void k_sumexp(const float* __restrict__ s, const float* __restrict__ gmax,
                         float* __restrict__ Z, int E) {
  __shared__ float wred[8];
  long long e = (long long)blockIdx.x * blockDim.x + threadIdx.x;
  float gm = *gmax;
  float v = 0.f;
  if (e < E) v = expf(s[e] - gm);
#pragma unroll
  for (int off = 16; off > 0; off >>= 1) v += __shfl_xor(v, off, 32);
  if ((threadIdx.x & 31) == 0) wred[threadIdx.x >> 5] = v;
  __syncthreads();
  if (threadIdx.x == 0) {
    float t = 0.f;
#pragma unroll
    for (int i = 0; i < 8; ++i) t += wred[i];
    atomicAdd(Z, t);
  }
}

// ---------------------------------------------------------------------------
// stage-2 scatter: out[d,:] += alpha[e] * V[m,:]   (wave per edge)
// ---------------------------------------------------------------------------
__global__ void k_scatter2(const int* __restrict__ ei2, const float* __restrict__ s,
                           const float* __restrict__ gmax, const float* __restrict__ Z,
                           const float* __restrict__ V, float* __restrict__ out, int E) {
  int lane = threadIdx.x & 31;
  long long w = ((long long)blockIdx.x * blockDim.x + threadIdx.x) >> 5;
  if (w >= E) return;                    // uniform per wave
  int e = (int)w;
  int m = ei2[e];
  int d = ei2[E + e];
  float a = expf(s[e] - *gmax) / *Z;     // broadcast loads, L2/L0 cached
  const float4 vv = ((const float4*)(V + (long long)m * C128))[lane];
  float* op = out + (long long)d * C128 + lane * 4;
  atomicAdd(op + 0, a * vv.x);
  atomicAdd(op + 1, a * vv.y);
  atomicAdd(op + 2, a * vv.z);
  atomicAdd(op + 3, a * vv.w);
}

// ---------------------------------------------------------------------------
// host-side orchestration
// ---------------------------------------------------------------------------
extern "C" void kernel_launch(void* const* d_in, const int* in_sizes, int n_in,
                              void* d_out, int out_size, void* d_ws, size_t ws_size,
                              hipStream_t stream) {
  (void)n_in; (void)ws_size;
  const float* x_src = (const float*)d_in[0];
  const float* x_mid = (const float*)d_in[1];
  const float* x_dst = (const float*)d_in[2];
  const int*   ei1   = (const int*)d_in[3];
  const int*   ei2   = (const int*)d_in[4];
  const float* W1w   = (const float*)d_in[5];
  const float* W1b   = (const float*)d_in[6];
  const float* W2w   = (const float*)d_in[7];
  const float* W2b   = (const float*)d_in[8];
  const float* qw    = (const float*)d_in[9];
  const float* qb    = (const float*)d_in[10];
  const float* kw    = (const float*)d_in[11];
  const float* kb    = (const float*)d_in[12];
  const float* vw    = (const float*)d_in[13];
  const float* vb    = (const float*)d_in[14];

  const int N_mid = in_sizes[1] / C128;
  const int N_dst = in_sizes[2] / C128;
  const int E1    = in_sizes[3] / 2;
  const int E2    = in_sizes[4] / 2;

  // workspace carve-out (256B aligned slices)
  char* wp = (char*)d_ws;
  auto take = [&](size_t bytes) { char* p = wp; wp += (bytes + 255) & ~(size_t)255; return p; };
  __bf16* W1bf = (__bf16*)take((size_t)C128 * C128 * 2);
  __bf16* W2bf = (__bf16*)take((size_t)C128 * C128 * 2);
  __bf16* Kwbf = (__bf16*)take((size_t)C128 * C128 * 2);
  __bf16* Vwbf = (__bf16*)take((size_t)C128 * C128 * 2);
  __bf16* Qwbf = (__bf16*)take((size_t)C128 * C128 * 2);
  float*  bsum = (float*)take(C128 * sizeof(float));
  float*  gmax = (float*)take(sizeof(float));
  float*  Z    = (float*)take(sizeof(float));
  float*  agg  = (float*)take((size_t)N_mid * C128 * sizeof(float));
  float*  deg  = (float*)take((size_t)N_mid * sizeof(float));
  float*  Hmid = (float*)take((size_t)N_mid * C128 * sizeof(float));
  float*  Kmat = (float*)take((size_t)N_mid * C128 * sizeof(float));
  float*  Vmat = (float*)take((size_t)N_mid * C128 * sizeof(float));
  float*  Qmat = (float*)take((size_t)N_dst * C128 * sizeof(float));
  float*  smat = (float*)take((size_t)E2 * sizeof(float));
  float*  outF = (float*)d_out;

  const int TB = 256;
  auto blk1 = [&](long long n) { return (unsigned)((n + TB - 1) / TB); };
  auto blk4 = [&](long long n) { long long q = (n + 3) >> 2; return (unsigned)((q + TB - 1) / TB); };

  // 1) init accumulators + fused bias + softmax state
  hipLaunchKernelGGL(k_init_small, dim3(1), dim3(128), 0, stream, W1b, W2b, bsum, gmax, Z);
  hipLaunchKernelGGL(k_zero_f32, dim3(blk4((long long)N_mid * C128)), dim3(TB), 0, stream,
                     agg, (long long)N_mid * C128);
  hipLaunchKernelGGL(k_zero_f32, dim3(blk4(N_mid)), dim3(TB), 0, stream, deg, (long long)N_mid);
  hipLaunchKernelGGL(k_zero_f32, dim3(blk4(out_size)), dim3(TB), 0, stream,
                     outF, (long long)out_size);

  // 2) weight conversion fp32 -> bf16 (row-major [c][j])
  const int WN = C128 * C128;
  hipLaunchKernelGGL(k_cvt_bf16, dim3(blk1(WN)), dim3(TB), 0, stream, W1w, W1bf, WN);
  hipLaunchKernelGGL(k_cvt_bf16, dim3(blk1(WN)), dim3(TB), 0, stream, W2w, W2bf, WN);
  hipLaunchKernelGGL(k_cvt_bf16, dim3(blk1(WN)), dim3(TB), 0, stream, kw,  Kwbf, WN);
  hipLaunchKernelGGL(k_cvt_bf16, dim3(blk1(WN)), dim3(TB), 0, stream, vw,  Vwbf, WN);
  hipLaunchKernelGGL(k_cvt_bf16, dim3(blk1(WN)), dim3(TB), 0, stream, qw,  Qwbf, WN);

  // 3) stage-1 edge scatter (wave per edge)
  hipLaunchKernelGGL(k_scatter1, dim3(blk1((long long)E1 * 32)), dim3(TB), 0, stream,
                     ei1, x_src, agg, deg, E1);

  // 4) node GEMMs via WMMA (128 rows / block)
  unsigned gmid = (unsigned)((N_mid + 127) / 128);
  unsigned gdst = (unsigned)((N_dst + 127) / 128);
  hipLaunchKernelGGL(k_gemm_mid, dim3(gmid), dim3(TB), 0, stream,
                     x_mid, agg, W1bf, W2bf, deg, bsum, Hmid, N_mid);
  hipLaunchKernelGGL(k_gemm_bias, dim3(gmid), dim3(TB), 0, stream, Hmid, Kwbf, kb, Kmat, N_mid);
  hipLaunchKernelGGL(k_gemm_bias, dim3(gmid), dim3(TB), 0, stream, Hmid, Vwbf, vb, Vmat, N_mid);
  hipLaunchKernelGGL(k_gemm_bias, dim3(gdst), dim3(TB), 0, stream, x_dst, Qwbf, qb, Qmat, N_dst);

  // 5) global softmax over E2 edges
  hipLaunchKernelGGL(k_scores, dim3(blk1(E2)), dim3(TB), 0, stream,
                     ei2, Qmat, Kmat, smat, gmax, E2);
  hipLaunchKernelGGL(k_sumexp, dim3(blk1(E2)), dim3(TB), 0, stream, smat, gmax, Z, E2);

  // 6) weighted scatter into d_out (wave per edge)
  hipLaunchKernelGGL(k_scatter2, dim3(blk1((long long)E2 * 32)), dim3(TB), 0, stream,
                     ei2, smat, gmax, Z, Vmat, outF, E2);
}
